// SurgeryAttention_24266565222468
// MI455X (gfx1250) — matrix operand order repaired and checked by reading, not compiled, not run on hardware
//
#include <hip/hip_runtime.h>
#include <stdint.h>

// ---------------------------------------------------------------------------
// Types
// ---------------------------------------------------------------------------
typedef __bf16 bf16;
typedef bf16  v16bf __attribute__((ext_vector_type(16)));
typedef bf16  v8bf  __attribute__((ext_vector_type(8)));
typedef bf16  v4bf  __attribute__((ext_vector_type(4)));
typedef float v8f   __attribute__((ext_vector_type(8)));

#define DIM       1024
#define SEQ       2048
#define BATCH     2
#define HEADS     16
#define HDIM      64
#define MTOT      (BATCH * SEQ)   // 4096
#define ATTN_SCALE 0.125f         // 1/sqrt(64)

// ---------------------------------------------------------------------------
// CDNA5 primitives
// ---------------------------------------------------------------------------
__device__ __forceinline__ v8f wmma_bf16(v16bf a, v16bf b, v8f c) {
  return __builtin_amdgcn_wmma_f32_16x16x32_bf16(
      /*neg_a=*/false, a, /*neg_b=*/false, b,
      /*c_mod=*/(short)0, c, /*reuse_a=*/false, /*reuse_b=*/false);
}

// Async global->LDS copy, 16 bytes per lane, tracked by ASYNCcnt.
// VDST = LDS byte address (low 32 bits of generic shared pointer),
// VADDR = 64-bit global address, saddr = off (GV mode).
__device__ __forceinline__ void async_copy_b128(void* lds_dst, const void* gsrc) {
  uint32_t lds_off = (uint32_t)(uintptr_t)lds_dst;
  asm volatile("global_load_async_to_lds_b128 %0, %1, off"
               :: "v"(lds_off), "v"(gsrc) : "memory");
}

__device__ __forceinline__ void wait_async0() {
  asm volatile("s_wait_asynccnt 0x0" ::: "memory");
}

// A-fragment (16x32 bf16, row-major LDS tile, `stride` elements per row).
// ISA layout: lanes 0-15 -> rows, K = {hlf*8..+7, 16+hlf*8..+7}
__device__ __forceinline__ v16bf load_a_frag(const bf16* base, int stride, int lane) {
  const int m = lane & 15, hlf = lane >> 4;
  const bf16* p = base + m * stride + hlf * 8;
  v8bf lo = *(const v8bf*)(p);        // K = hlf*8 + 0..7
  v8bf hi = *(const v8bf*)(p + 16);   // K = 16 + hlf*8 + 0..7
  return __builtin_shufflevector(lo, hi, 0,1,2,3,4,5,6,7,8,9,10,11,12,13,14,15);
}

// B-fragment (32x16) from an N-major tile: tile[n][k], `stride` = K pitch.
// ISA layout: lane n = lane&15, K = {hlf*8..+7, 16+hlf*8..+7}
__device__ __forceinline__ v16bf load_b_frag(const bf16* base, int stride, int lane) {
  const int n = lane & 15, hlf = lane >> 4;
  const bf16* p = base + n * stride + hlf * 8;
  v8bf lo = *(const v8bf*)(p);
  v8bf hi = *(const v8bf*)(p + 16);
  return __builtin_shufflevector(lo, hi, 0,1,2,3,4,5,6,7,8,9,10,11,12,13,14,15);
}

// ---------------------------------------------------------------------------
// Kernel 0a: x (f32) -> bf16, vectorized 4/thread
// ---------------------------------------------------------------------------
__global__ __launch_bounds__(256)
void convert_x_kernel(const float* __restrict__ x, bf16* __restrict__ xb) {
  const size_t i4 = ((size_t)blockIdx.x * 256 + threadIdx.x) * 4;
  float4 f = *(const float4*)(x + i4);
  v4bf o;
  o[0] = (bf16)f.x; o[1] = (bf16)f.y; o[2] = (bf16)f.z; o[3] = (bf16)f.w;
  *(v4bf*)(xb + i4) = o;
}

// ---------------------------------------------------------------------------
// Kernel 0b: slice W_qkv's V-columns + transpose both weights into bf16 [N][K]
// ---------------------------------------------------------------------------
__global__ __launch_bounds__(256)
void prep_weights_kernel(const float* __restrict__ Wqkv,
                         const float* __restrict__ Wproj,
                         bf16* __restrict__ WvT, bf16* __restrict__ WpT) {
  const size_t idx = (size_t)blockIdx.x * 256 + threadIdx.x; // 1M threads
  const int k = (int)(idx & 1023);
  const int c = (int)(idx >> 10);
  WvT[(size_t)c * DIM + k] = (bf16)Wqkv[(size_t)k * (3 * DIM) + 2 * DIM + c];
  WpT[(size_t)c * DIM + k] = (bf16)Wproj[(size_t)k * DIM + c];
}

// ---------------------------------------------------------------------------
// GEMM: out[M,N] = A[M,K] @ BT[N,K]^T + bias   (A, BT bf16; acc f32)
// Block: 256 thr (8 waves). Tile 128x128, K-step 32. Wave -> 32x64 (8 WMMA).
// Double-buffered LDS filled by async global->LDS copies.
// ---------------------------------------------------------------------------
// Stage one 128x32 bf16 tile (row-major, stride 32) via async b128 copies.
__device__ __forceinline__ void stage_tile_async(bf16* dstLds, const bf16* gbase,
                                                 int K, int tid) {
  #pragma unroll
  for (int i = 0; i < 2; ++i) {
    const int c = tid + i * 256;          // 512 chunks of 16B
    const int row = c >> 2, col = (c & 3) * 8;
    async_copy_b128(&dstLds[row * 32 + col], gbase + (size_t)row * K + col);
  }
}

template <bool OUT_BF16>
__global__ __launch_bounds__(256)
void gemm_kernel(const bf16* __restrict__ A, const bf16* __restrict__ BT,
                 const float* __restrict__ bias, void* __restrict__ outp,
                 int M, int N, int K) {
  __shared__ alignas(64) bf16 As[2][128 * 32];   // 2 x 8 KB
  __shared__ alignas(64) bf16 Bs[2][128 * 32];   // 2 x 8 KB, [n][k]

  const int tid = threadIdx.x;
  const int lane = tid & 31, w = tid >> 5;
  const int wm = (w & 3) * 32;     // wave M offset in tile (4 waves over 128)
  const int wn = (w >> 2) * 64;    // wave N offset in tile (2 waves over 128)
  const int bm = blockIdx.y * 128, bn = blockIdx.x * 128;

  const bf16* Ag = A + (size_t)bm * K;
  const bf16* Bg = BT + (size_t)bn * K;

  v8f acc[2][4] = {};
  const int nk = K >> 5;

  // prologue: fill buffer 0
  stage_tile_async(As[0], Ag, K, tid);
  stage_tile_async(Bs[0], Bg, K, tid);

  for (int kt = 0; kt < nk; ++kt) {
    const int cur = kt & 1;
    wait_async0();        // this wave's copies for tile kt have landed
    __syncthreads();      // ... and everyone else's; prev buffer free for reuse

    if (kt + 1 < nk) {    // kick off next tile while we compute this one
      const int k1 = (kt + 1) << 5;
      stage_tile_async(As[cur ^ 1], Ag + k1, K, tid);
      stage_tile_async(Bs[cur ^ 1], Bg + k1, K, tid);
    }

    v16bf aF[2], bF[4];
    #pragma unroll
    for (int mi = 0; mi < 2; ++mi)
      aF[mi] = load_a_frag(&As[cur][(wm + mi * 16) * 32], 32, lane);
    #pragma unroll
    for (int ni = 0; ni < 4; ++ni)
      bF[ni] = load_b_frag(&Bs[cur][(wn + ni * 16) * 32], 32, lane);

    #pragma unroll
    for (int mi = 0; mi < 2; ++mi)
      #pragma unroll
      for (int ni = 0; ni < 4; ++ni)
        acc[mi][ni] = wmma_bf16(aF[mi], bF[ni], acc[mi][ni]);
  }

  // epilogue: C layout = lane column (lane&15), rows (lane>>4)*8 + r
  const int hlf = lane >> 4, nn = lane & 15;
  #pragma unroll
  for (int mi = 0; mi < 2; ++mi) {
    #pragma unroll
    for (int ni = 0; ni < 4; ++ni) {
      const int col = bn + wn + ni * 16 + nn;
      const float bv = bias[col];
      #pragma unroll
      for (int r = 0; r < 8; ++r) {
        const int row = bm + wm + mi * 16 + hlf * 8 + r;
        const float v = acc[mi][ni][r] + bv;
        if (OUT_BF16)
          ((bf16*)outp)[(size_t)row * N + col] = (bf16)v;
        else
          ((float*)outp)[(size_t)row * N + col] = v;
      }
    }
  }
}

// ---------------------------------------------------------------------------
// Flash attention with Q=K=V (the reference quirk). One block = (b,h) x 128 rows.
// 8 waves, each wave owns 16 query rows. Key blocks of 64, hd = 64.
// ---------------------------------------------------------------------------
__global__ __launch_bounds__(256)
void attn_kernel(const bf16* __restrict__ V, bf16* __restrict__ O) {
  __shared__ alignas(64) bf16 Qs[128 * 64];     // 16 KB
  __shared__ alignas(64) bf16 Ks[64 * 64];      // 8 KB  [key][hd]
  __shared__ alignas(64) bf16 VTs[64 * 64];     // 8 KB  [hd][key]
  __shared__ alignas(64) bf16 Ps[8][16 * 64];   // 16 KB per-wave P scratch

  const int tid = threadIdx.x, lane = tid & 31, w = tid >> 5;
  const int hlf = lane >> 4, ln = lane & 15;
  const int bh = blockIdx.y;
  const int b = bh >> 4, h = bh & 15;
  const int q0 = blockIdx.x * 128;
  const bf16* Vbase = V + (size_t)b * SEQ * DIM + h * HDIM;

  // ---- async-load Q tile (128 x 64 bf16 = 16 KB; 4x16B per thread) ----
  #pragma unroll
  for (int i = 0; i < 4; ++i) {
    const int c = tid + i * 256;          // 1024 chunks of 16B
    const int row = c >> 3, col = (c & 7) * 8;
    async_copy_b128(&Qs[row * 64 + col], Vbase + (size_t)(q0 + row) * DIM + col);
  }
  wait_async0();   // each wave waits for its own Q copies; barrier below orders all

  float rmax[8], rsum[8];
  v8f accO[4] = {};
  #pragma unroll
  for (int r = 0; r < 8; ++r) { rmax[r] = -1e30f; rsum[r] = 0.0f; }

  const int wrow = w * 16;   // wave's query rows within tile

  for (int kb = 0; kb < SEQ / 64; ++kb) {
    // ---- stage K/V tile (64 keys x 64 hd) + transposed copy (for P@V B-frags)
    {
      const int j = tid >> 2, seg = (tid & 3) * 16;
      const bf16* g = Vbase + (size_t)(kb * 64 + j) * DIM + seg;
      v8bf d0 = *(const v8bf*)(g);
      v8bf d1 = *(const v8bf*)(g + 8);
      *(v8bf*)&Ks[j * 64 + seg]     = d0;
      *(v8bf*)&Ks[j * 64 + seg + 8] = d1;
      #pragma unroll
      for (int e = 0; e < 8; ++e) {
        VTs[(seg + e) * 64 + j]     = d0[e];
        VTs[(seg + 8 + e) * 64 + j] = d1[e];
      }
    }
    __syncthreads();

    // ---- S = Q @ K^T (16 rows x 64 keys per wave) ----
    v8f accS[4] = {};
    #pragma unroll
    for (int kc = 0; kc < 2; ++kc) {
      v16bf aQ = load_a_frag(&Qs[wrow * 64 + kc * 32], 64, lane);
      #pragma unroll
      for (int ni = 0; ni < 4; ++ni) {
        v16bf bK = load_b_frag(&Ks[(ni * 16) * 64 + kc * 32], 64, lane);
        accS[ni] = wmma_bf16(aQ, bK, accS[ni]);
      }
    }

    // ---- online softmax (row = hlf*8 + r; reduce over 16 lanes in half) ----
    #pragma unroll
    for (int r = 0; r < 8; ++r) {
      float s0 = accS[0][r] * ATTN_SCALE;
      float s1 = accS[1][r] * ATTN_SCALE;
      float s2 = accS[2][r] * ATTN_SCALE;
      float s3 = accS[3][r] * ATTN_SCALE;
      float mx = fmaxf(fmaxf(s0, s1), fmaxf(s2, s3));
      mx = fmaxf(mx, __shfl_xor(mx, 1));
      mx = fmaxf(mx, __shfl_xor(mx, 2));
      mx = fmaxf(mx, __shfl_xor(mx, 4));
      mx = fmaxf(mx, __shfl_xor(mx, 8));
      const float mnew = fmaxf(rmax[r], mx);
      const float f = __expf(rmax[r] - mnew);
      const float p0 = __expf(s0 - mnew);
      const float p1 = __expf(s1 - mnew);
      const float p2 = __expf(s2 - mnew);
      const float p3 = __expf(s3 - mnew);
      float ps = p0 + p1 + p2 + p3;
      ps += __shfl_xor(ps, 1);
      ps += __shfl_xor(ps, 2);
      ps += __shfl_xor(ps, 4);
      ps += __shfl_xor(ps, 8);
      rsum[r] = rsum[r] * f + ps;
      rmax[r] = mnew;
      #pragma unroll
      for (int ci = 0; ci < 4; ++ci) accO[ci][r] *= f;
      bf16* pr = &Ps[w][(hlf * 8 + r) * 64 + ln];
      pr[0]  = (bf16)p0;
      pr[16] = (bf16)p1;
      pr[32] = (bf16)p2;
      pr[48] = (bf16)p3;
    }
    // per-wave LDS writes/reads stay in order; Ps has no cross-wave dependency

    // ---- O += P @ V ----
    #pragma unroll
    for (int kc = 0; kc < 2; ++kc) {
      v16bf aP = load_a_frag(&Ps[w][kc * 32], 64, lane);
      #pragma unroll
      for (int ci = 0; ci < 4; ++ci) {
        v16bf bV = load_b_frag(&VTs[(ci * 16) * 64 + kc * 32], 64, lane);
        accO[ci] = wmma_bf16(aP, bV, accO[ci]);
      }
    }
    __syncthreads();   // protect Ks/VTs before next tile load
  }

  // ---- normalize + store attn output (bf16, [B*N, DIM] with head offset) ----
  #pragma unroll
  for (int r = 0; r < 8; ++r) {
    const float inv = 1.0f / rsum[r];
    const int row = q0 + wrow + hlf * 8 + r;
    bf16* op = O + ((size_t)b * SEQ + row) * DIM + h * HDIM + ln;
    op[0]  = (bf16)(accO[0][r] * inv);
    op[16] = (bf16)(accO[1][r] * inv);
    op[32] = (bf16)(accO[2][r] * inv);
    op[48] = (bf16)(accO[3][r] * inv);
  }
}

// ---------------------------------------------------------------------------
// Launch
// ---------------------------------------------------------------------------
extern "C" void kernel_launch(void* const* d_in, const int* in_sizes, int n_in,
                              void* d_out, int out_size, void* d_ws, size_t ws_size,
                              hipStream_t stream) {
  (void)in_sizes; (void)n_in; (void)out_size; (void)ws_size;
  const float* x      = (const float*)d_in[0];
  const float* W_qkv  = (const float*)d_in[1];
  const float* b_qkv  = (const float*)d_in[2];
  const float* W_proj = (const float*)d_in[3];
  const float* b_proj = (const float*)d_in[4];
  float* out = (float*)d_out;

  char* ws = (char*)d_ws;
  bf16* WvT = (bf16*)(ws);                   //  2 MB: W_v^T   [DIM][DIM]
  bf16* WpT = (bf16*)(ws + (2ull << 20));    //  2 MB: W_p^T   [DIM][DIM]
  bf16* Xb  = (bf16*)(ws + (4ull << 20));    //  8 MB: x bf16  [MTOT][DIM]
  bf16* Vb  = (bf16*)(ws + (12ull << 20));   //  8 MB: V       [MTOT][DIM]
  bf16* AOb = (bf16*)(ws + (20ull << 20));   //  8 MB: attnOut [MTOT][DIM]

  // 0) conversions
  convert_x_kernel<<<dim3((MTOT * DIM) / (256 * 4)), dim3(256), 0, stream>>>(x, Xb);
  prep_weights_kernel<<<dim3((DIM * DIM) / 256), dim3(256), 0, stream>>>(
      W_qkv, W_proj, WvT, WpT);

  // 1) V = x @ W_v + b_v   (bf16 in, bf16 out)
  gemm_kernel<true><<<dim3(DIM / 128, MTOT / 128), dim3(256), 0, stream>>>(
      Xb, WvT, b_qkv + 2 * DIM, (void*)Vb, MTOT, DIM, DIM);

  // 2) flash attention (Q=K=V)
  attn_kernel<<<dim3(SEQ / 128, BATCH * HEADS), dim3(256), 0, stream>>>(Vb, AOb);

  // 3) out = attnOut @ W_proj + b_proj   (bf16 in, f32 out)
  gemm_kernel<false><<<dim3(DIM / 128, MTOT / 128), dim3(256), 0, stream>>>(
      AOb, WpT, b_proj, (void*)out, MTOT, DIM, DIM);
}
